// full_model_86784109183053
// MI455X (gfx1250) — compile-verified, matrix operand order checked by reading, not run on hardware
//
#include <hip/hip_runtime.h>
#include <hip/hip_bf16.h>
#include <math.h>

typedef _Float16 h16;
typedef __attribute__((ext_vector_type(16))) _Float16 v16h;
typedef __attribute__((ext_vector_type(8)))  _Float16 v8h;
typedef __attribute__((ext_vector_type(8)))  float    v8f;
typedef int v4i_ __attribute__((vector_size(4 * sizeof(int))));

#define FHD   31
#define NAA   9
#define KTOP  6000
#define NPROP 1500
#define CHN   512

// ---- CDNA5 async global->LDS staging (guarded; falls back to plain LDS stores) ----
#if defined(__has_builtin)
#if __has_builtin(__builtin_amdgcn_global_load_async_to_lds_b128) && \
    __has_builtin(__builtin_amdgcn_s_wait_asynccnt)
#define USE_ASYNC_LDS 1
#endif
#endif

__device__ __forceinline__ void stage16B(const h16* gsrc, h16* ldst) {
#ifdef USE_ASYNC_LDS
  __builtin_amdgcn_global_load_async_to_lds_b128(
      (__attribute__((address_space(1))) v4i_*)(uintptr_t)gsrc,
      (__attribute__((address_space(3))) v4i_*)ldst, 0, 0);
#else
  *(v8h*)ldst = *(const v8h*)gsrc;
#endif
}

__device__ __forceinline__ void stage_wait() {
#ifdef USE_ASYNC_LDS
  __builtin_amdgcn_s_wait_asynccnt(0);
#endif
}

// ---------------- f32 -> f16 convert ----------------
__global__ void cvt_f32_f16_kernel(const float* __restrict__ src, h16* __restrict__ dst, long n) {
  long i = (long)blockIdx.x * blockDim.x + threadIdx.x;
  long stride = (long)gridDim.x * blockDim.x;
  for (; i < n; i += stride) dst[i] = (h16)src[i];
}

// ---------------- f32 [K][N] -> f16 [N][K] transpose-convert ----------------
__global__ void cvt_transpose_kernel(const float* __restrict__ src, h16* __restrict__ dst,
                                     int K, int N) {
  long total = (long)K * N;
  long i = (long)blockIdx.x * blockDim.x + threadIdx.x;
  long stride = (long)gridDim.x * blockDim.x;
  for (; i < total; i += stride) {
    int k = (int)(i / N);
    int n = (int)(i % N);
    dst[(size_t)n * K + k] = (h16)src[i];
  }
}

// ---------------- first conv (Cin=3), direct, relu ----------------
__global__ void conv3x3_c3_kernel(const float* __restrict__ x, const float* __restrict__ W,
                                  const float* __restrict__ b, h16* __restrict__ out, int H) {
  long total = (long)H * H * 64;
  long i = (long)blockIdx.x * blockDim.x + threadIdx.x;
  long stride = (long)gridDim.x * blockDim.x;
  for (; i < total; i += stride) {
    int co = (int)(i & 63);
    int p  = (int)(i >> 6);
    int px = p % H, py = p / H;
    float acc = b[co];
    for (int dy = 0; dy < 3; ++dy) {
      int yy = py + dy - 1;
      if (yy < 0 || yy >= H) continue;
      for (int dx = 0; dx < 3; ++dx) {
        int xx = px + dx - 1;
        if (xx < 0 || xx >= H) continue;
        const float* xin = x + ((size_t)yy * H + xx) * 3;
        const float* wp  = W + ((size_t)(dy * 3 + dx) * 3) * 64 + co;
        acc += xin[0] * wp[0] + xin[1] * wp[64] + xin[2] * wp[128];
      }
    }
    out[i] = (h16)fmaxf(acc, 0.f);
  }
}

// ---------------- 2x2 maxpool (f16) ----------------
__global__ void maxpool2_f16_kernel(const h16* __restrict__ in, h16* __restrict__ out,
                                    int Hin, int C) {
  int Hout = Hin >> 1;
  long total = (long)Hout * Hout * C;
  long i = (long)blockIdx.x * blockDim.x + threadIdx.x;
  long stride = (long)gridDim.x * blockDim.x;
  for (; i < total; i += stride) {
    int c = (int)(i % C);
    int p = (int)(i / C);
    int ox = p % Hout, oy = p / Hout;
    float m = -INFINITY;
    for (int sy = 0; sy < 2; ++sy)
      for (int sx = 0; sx < 2; ++sx) {
        float v = (float)in[((size_t)(2 * oy + sy) * Hin + (2 * ox + sx)) * C + c];
        m = fmaxf(m, v);
      }
    out[i] = (h16)m;
  }
}

// ---------------- shared WMMA micro-kernel pieces ----------------
// Block: 256 threads = 8 wave32, tile 64(M) x 128(N), K step 32, double-buffered LDS.
// Wave tile 32x32: 2 A-frags x 2 B-frags -> 4 WMMAs per K-step.
#define LDSA_STRIDE 40
#define LDSB_STRIDE 40

__device__ __forceinline__ v16h load_frag(const h16* base, int rowstride, int r0, int lane) {
  v16h f;
  const h16* p = base + (size_t)(r0 + (lane & 15)) * rowstride + ((lane >> 4) * 16);
  #pragma unroll
  for (int q = 0; q < 16; ++q) f[q] = p[q];
  return f;
}

// ---------------- generic WMMA GEMM: C = act(A[M,K] * Bt[N,K]^T + bias) ----------------
// Bt is the K-major (transposed) weight matrix: Bt[n][k].
__launch_bounds__(256)
__global__ void wmma_gemm_kernel(const h16* __restrict__ A, const h16* __restrict__ Bt,
                                 const float* __restrict__ bias, void* __restrict__ outp,
                                 int M, int N, int K, int do_relu, int out_f16) {
  __shared__ h16 lds_a[2][64][LDSA_STRIDE];
  __shared__ h16 lds_bT[2][128][LDSB_STRIDE];
  const int tid  = threadIdx.x;
  const int lane = tid & 31;
  const int wave = tid >> 5;
  const int wm = wave & 1;          // 2 wave-rows  (32 M each)
  const int wn = wave >> 1;         // 4 wave-cols  (32 N each)
  const int m0 = blockIdx.x * 64;
  const int n0 = blockIdx.y * 128;
  const int a_m = tid >> 2;         // 0..63
  const int a_k = (tid & 3) * 8;    // 0,8,16,24
  const int b_n = tid >> 1;         // 0..127
  const int b_k = (tid & 1) * 16;   // 0,16

  auto stageA = [&](int buf, int k0) {
    int gm = m0 + a_m;
    if (gm < M) {
      const h16* src = A + (size_t)gm * K + (k0 + a_k);
      __builtin_prefetch(src + 32, 0, 1);
      stage16B(src, &lds_a[buf][a_m][a_k]);
    } else {
      v8h z = {};
      *(v8h*)&lds_a[buf][a_m][a_k] = z;
    }
  };
  auto stageB = [&](int buf, int k0) {
    int gn = n0 + b_n;
    if (gn < N) {
      const h16* src = Bt + (size_t)gn * K + (k0 + b_k);
      stage16B(src,     &lds_bT[buf][b_n][b_k]);
      stage16B(src + 8, &lds_bT[buf][b_n][b_k + 8]);
    } else {
      v8h z = {};
      *(v8h*)&lds_bT[buf][b_n][b_k] = z;
      *(v8h*)&lds_bT[buf][b_n][b_k + 8] = z;
    }
  };

  v8f c00 = {}, c01 = {}, c10 = {}, c11 = {};
  stageA(0, 0);
  stageB(0, 0);
  stage_wait();
  __syncthreads();
  int buf = 0;
  for (int k0 = 0; k0 < K; k0 += 32) {
    if (k0 + 32 < K) {            // prefetch next tile into the other buffer
      stageA(buf ^ 1, k0 + 32);
      stageB(buf ^ 1, k0 + 32);
    }
    v16h a0 = load_frag(&lds_a[buf][0][0],  LDSA_STRIDE, wm * 32,      lane);
    v16h a1 = load_frag(&lds_a[buf][0][0],  LDSA_STRIDE, wm * 32 + 16, lane);
    v16h b0 = load_frag(&lds_bT[buf][0][0], LDSB_STRIDE, wn * 32,      lane);
    v16h b1 = load_frag(&lds_bT[buf][0][0], LDSB_STRIDE, wn * 32 + 16, lane);
    c00 = __builtin_amdgcn_wmma_f32_16x16x32_f16(false, a0, false, b0, (short)0, c00, false, false);
    c01 = __builtin_amdgcn_wmma_f32_16x16x32_f16(false, a0, false, b1, (short)0, c01, false, false);
    c10 = __builtin_amdgcn_wmma_f32_16x16x32_f16(false, a1, false, b0, (short)0, c10, false, false);
    c11 = __builtin_amdgcn_wmma_f32_16x16x32_f16(false, a1, false, b1, (short)0, c11, false, false);
    stage_wait();
    __syncthreads();
    buf ^= 1;
  }
  // epilogue
  const v8f accs[4] = {c00, c01, c10, c11};
  #pragma unroll
  for (int mi = 0; mi < 2; ++mi) {
    #pragma unroll
    for (int ni = 0; ni < 2; ++ni) {
      const v8f acc = accs[mi * 2 + ni];
      int gn = n0 + wn * 32 + ni * 16 + (lane & 15);
      if (gn < N) {
        float bv = bias ? bias[gn] : 0.f;
        #pragma unroll
        for (int r = 0; r < 8; ++r) {
          int gm = m0 + wm * 32 + mi * 16 + r + ((lane >> 4) << 3);
          if (gm < M) {
            float v = acc[r] + bv;
            if (do_relu) v = fmaxf(v, 0.f);
            if (out_f16) ((h16*)outp)[(size_t)gm * N + gn] = (h16)v;
            else         ((float*)outp)[(size_t)gm * N + gn] = v;
          }
        }
      }
    }
  }
}

// ---------------- 3x3 SAME conv via implicit-im2col WMMA, relu, f16 out ----------------
// Wt is transposed: Wt[co][k], k = (dy*3+dx)*Cin + ci.
__launch_bounds__(256)
__global__ void conv3x3_wmma_kernel(const h16* __restrict__ in, const h16* __restrict__ Wt,
                                    const float* __restrict__ bias, h16* __restrict__ out,
                                    int H, int Cin, int Cout) {
  __shared__ h16 lds_a[2][64][LDSA_STRIDE];
  __shared__ h16 lds_bT[2][128][LDSB_STRIDE];
  const int M = H * H;
  const int K = 9 * Cin;
  const int tid  = threadIdx.x;
  const int lane = tid & 31;
  const int wave = tid >> 5;
  const int wm = wave & 1, wn = wave >> 1;
  const int m0 = blockIdx.x * 64;
  const int n0 = blockIdx.y * 128;
  const int a_m = tid >> 2;
  const int a_k = (tid & 3) * 8;
  const int b_n = tid >> 1;
  const int b_k = (tid & 1) * 16;

  auto stageA = [&](int buf, int k0) {
    // implicit im2col: 8-half chunk never crosses a tap (Cin multiple of 32)
    int gm = m0 + a_m;
    int k  = k0 + a_k;
    int tap = k / Cin;
    int ci  = k - tap * Cin;
    int dy = tap / 3 - 1, dx = tap % 3 - 1;
    bool ok = false;
    const h16* src = in;
    if (gm < M) {
      int py = gm / H, px = gm % H;
      int yy = py + dy, xx = px + dx;
      if (yy >= 0 && yy < H && xx >= 0 && xx < H) {
        src = in + ((size_t)yy * H + xx) * Cin + ci;
        ok = true;
      }
    }
    if (ok) {
      stage16B(src, &lds_a[buf][a_m][a_k]);
    } else {
      v8h z = {};
      *(v8h*)&lds_a[buf][a_m][a_k] = z;
    }
  };
  auto stageB = [&](int buf, int k0) {
    int gn = n0 + b_n;
    if (gn < Cout) {
      const h16* src = Wt + (size_t)gn * K + (k0 + b_k);
      stage16B(src,     &lds_bT[buf][b_n][b_k]);
      stage16B(src + 8, &lds_bT[buf][b_n][b_k + 8]);
    } else {
      v8h z = {};
      *(v8h*)&lds_bT[buf][b_n][b_k] = z;
      *(v8h*)&lds_bT[buf][b_n][b_k + 8] = z;
    }
  };

  v8f c00 = {}, c01 = {}, c10 = {}, c11 = {};
  stageA(0, 0);
  stageB(0, 0);
  stage_wait();
  __syncthreads();
  int buf = 0;
  for (int k0 = 0; k0 < K; k0 += 32) {
    if (k0 + 32 < K) {
      stageA(buf ^ 1, k0 + 32);
      stageB(buf ^ 1, k0 + 32);
    }
    v16h a0 = load_frag(&lds_a[buf][0][0],  LDSA_STRIDE, wm * 32,      lane);
    v16h a1 = load_frag(&lds_a[buf][0][0],  LDSA_STRIDE, wm * 32 + 16, lane);
    v16h b0 = load_frag(&lds_bT[buf][0][0], LDSB_STRIDE, wn * 32,      lane);
    v16h b1 = load_frag(&lds_bT[buf][0][0], LDSB_STRIDE, wn * 32 + 16, lane);
    c00 = __builtin_amdgcn_wmma_f32_16x16x32_f16(false, a0, false, b0, (short)0, c00, false, false);
    c01 = __builtin_amdgcn_wmma_f32_16x16x32_f16(false, a0, false, b1, (short)0, c01, false, false);
    c10 = __builtin_amdgcn_wmma_f32_16x16x32_f16(false, a1, false, b0, (short)0, c10, false, false);
    c11 = __builtin_amdgcn_wmma_f32_16x16x32_f16(false, a1, false, b1, (short)0, c11, false, false);
    stage_wait();
    __syncthreads();
    buf ^= 1;
  }
  const v8f accs[4] = {c00, c01, c10, c11};
  #pragma unroll
  for (int mi = 0; mi < 2; ++mi) {
    #pragma unroll
    for (int ni = 0; ni < 2; ++ni) {
      const v8f acc = accs[mi * 2 + ni];
      int gn = n0 + wn * 32 + ni * 16 + (lane & 15);
      if (gn < Cout) {
        const float bv = bias[gn];
        #pragma unroll
        for (int r = 0; r < 8; ++r) {
          int gm = m0 + wm * 32 + mi * 16 + r + ((lane >> 4) << 3);
          if (gm < M) {
            float v = fmaxf(acc[r] + bv, 0.f);
            out[(size_t)gm * Cout + gn] = (h16)v;
          }
        }
      }
    }
  }
}

// ---------------- small GEMM (1x1 RPN heads), f16 A x f32 W, optional sigmoid ----------------
__global__ void small_gemm_kernel(const h16* __restrict__ A, const float* __restrict__ W,
                                  const float* __restrict__ b, float* __restrict__ out,
                                  int M, int K, int N, int act) {
  long total = (long)M * N;
  long i = (long)blockIdx.x * blockDim.x + threadIdx.x;
  long stride = (long)gridDim.x * blockDim.x;
  for (; i < total; i += stride) {
    int n = (int)(i % N);
    int m = (int)(i / N);
    float acc = b[n];
    const h16* ap = A + (size_t)m * K;
    for (int k = 0; k < K; ++k) acc += (float)ap[k] * W[(size_t)k * N + n];
    if (act == 1) acc = 1.f / (1.f + __expf(-acc));
    out[i] = acc;
  }
}

// ---------------- top-k by exhaustive rank (total order, tie-break by index) ----------------
__global__ void topk_rank_kernel(const float* __restrict__ s, int* __restrict__ rank, int n) {
  int i = blockIdx.x * blockDim.x + threadIdx.x;
  if (i >= n) return;
  float si = s[i];
  int r = 0;
  for (int j = 0; j < n; ++j) {
    float sj = s[j];
    r += (sj > si) || (sj == si && j < i);
  }
  rank[i] = r;
}

// ---------------- decode anchors + deltas, scatter into score order ----------------
__global__ void decode_scatter_kernel(const int* __restrict__ rank, const float* __restrict__ reg,
                                      const float* __restrict__ anchors, float* __restrict__ cand,
                                      int n) {
  int i = blockIdx.x * blockDim.x + threadIdx.x;
  if (i >= n) return;
  int r = rank[i];
  if (r >= KTOP) return;
  int a = i % NAA;
  int p = i / NAA;                 // p = y*FHD + x
  const float* anc = anchors + ((size_t)p * NAA + a) * 4;
  float ay = (anc[0] + anc[2]) * 0.5f, ax = (anc[1] + anc[3]) * 0.5f;
  float ah = anc[2] - anc[0], aw = anc[3] - anc[1];
  const float* d = reg + (size_t)p * (NAA * 4) + a * 4;
  float cy = ay + d[0] * ah;
  float cx = ax + d[1] * aw;
  float ph = ah * __expf(d[2]);
  float pw = aw * __expf(d[3]);
  cand[(size_t)r * 4 + 0] = cy - ph * 0.5f;
  cand[(size_t)r * 4 + 1] = cx - pw * 0.5f;
  cand[(size_t)r * 4 + 2] = cy + ph * 0.5f;
  cand[(size_t)r * 4 + 3] = cx + pw * 0.5f;
}

// ---------------- greedy NMS (single workgroup, flags+areas in LDS) ----------------
__launch_bounds__(1024)
__global__ void nms_kernel(const float* __restrict__ cand, float* __restrict__ prop) {
  __shared__ unsigned char active[KTOP];
  __shared__ float area[KTOP];
  const int tid = threadIdx.x;
  for (int i = tid; i < KTOP; i += 1024) {
    active[i] = 1;
    float y1 = cand[i * 4 + 0], x1 = cand[i * 4 + 1];
    float y2 = cand[i * 4 + 2], x2 = cand[i * 4 + 3];
    area[i] = (y2 - y1) * (x2 - x1);
  }
  for (int i = tid; i < NPROP * 4; i += 1024) prop[i] = 0.f;
  __syncthreads();
  for (int i = 0; i < KTOP - 1; ++i) {
    if (active[i]) {
      const float y1i = cand[i * 4 + 0], x1i = cand[i * 4 + 1];
      const float y2i = cand[i * 4 + 2], x2i = cand[i * 4 + 3];
      const float ai  = area[i];
      for (int j = i + 1 + tid; j < KTOP; j += 1024) {
        if (!active[j]) continue;
        float yy1 = fmaxf(y1i, cand[j * 4 + 0]);
        float xx1 = fmaxf(x1i, cand[j * 4 + 1]);
        float yy2 = fminf(y2i, cand[j * 4 + 2]);
        float xx2 = fminf(x2i, cand[j * 4 + 3]);
        float inter = fmaxf(yy2 - yy1, 0.f) * fmaxf(xx2 - xx1, 0.f);
        float iou = inter / (ai + area[j] - inter + 1e-8f);
        if (iou > 0.7f) active[j] = 0;
      }
    }
    __syncthreads();
  }
  if (tid == 0) {
    int cnt = 0;
    for (int i = 0; i < KTOP && cnt < NPROP; ++i) {
      if (active[i]) {
        #pragma unroll
        for (int c = 0; c < 4; ++c)
          prop[cnt * 4 + c] = fminf(fmaxf(cand[i * 4 + c], 0.f), 1.f);
        ++cnt;
      }
    }
  }
}

// ---------------- fused crop_and_resize(14x14) + 2x2 maxpool -> z[1500][7*7*512] f16 ----------
__launch_bounds__(256)
__global__ void crop_pool_kernel(const h16* __restrict__ fmap, const float* __restrict__ prop,
                                 h16* __restrict__ z) {
  const int b  = blockIdx.x;
  const int oy = blockIdx.y / 7, ox = blockIdx.y % 7;
  const float by1 = prop[b * 4 + 0], bx1 = prop[b * 4 + 1];
  const float by2 = prop[b * 4 + 2], bx2 = prop[b * 4 + 3];
  const int tid = threadIdx.x;
  float best0 = -INFINITY, best1 = -INFINITY;
  const int c0 = tid, c1 = tid + 256;
  #pragma unroll
  for (int s = 0; s < 4; ++s) {
    int cy = oy * 2 + (s >> 1);
    int cx = ox * 2 + (s & 1);
    float ty = (float)cy * (1.f / 13.f);
    float tx = (float)cx * (1.f / 13.f);
    float ys = (by1 + ty * (by2 - by1)) * 30.f;
    float xs = (bx1 + tx * (bx2 - bx1)) * 30.f;
    float y0f = floorf(ys), x0f = floorf(xs);
    float wy = ys - y0f, wx = xs - x0f;
    int y0 = min(max((int)y0f, 0), 30);
    int y1 = min(max((int)y0f + 1, 0), 30);
    int x0 = min(max((int)x0f, 0), 30);
    int x1 = min(max((int)x0f + 1, 0), 30);
    bool valid = (ys >= 0.f) && (ys <= 30.f) && (xs >= 0.f) && (xs <= 30.f);
    float w00 = (1.f - wy) * (1.f - wx), w01 = (1.f - wy) * wx;
    float w10 = wy * (1.f - wx),         w11 = wy * wx;
    const h16* p00 = fmap + ((size_t)y0 * FHD + x0) * CHN;
    const h16* p01 = fmap + ((size_t)y0 * FHD + x1) * CHN;
    const h16* p10 = fmap + ((size_t)y1 * FHD + x0) * CHN;
    const h16* p11 = fmap + ((size_t)y1 * FHD + x1) * CHN;
    float v0 = valid ? (w00 * (float)p00[c0] + w01 * (float)p01[c0] +
                        w10 * (float)p10[c0] + w11 * (float)p11[c0]) : 0.f;
    float v1 = valid ? (w00 * (float)p00[c1] + w01 * (float)p01[c1] +
                        w10 * (float)p10[c1] + w11 * (float)p11[c1]) : 0.f;
    best0 = fmaxf(best0, v0);
    best1 = fmaxf(best1, v1);
  }
  size_t base = ((size_t)b * 49 + blockIdx.y) * CHN;
  z[base + c0] = (h16)best0;
  z[base + c1] = (h16)best1;
}

// ---------------- softmax over 21 classes, one wave per row ----------------
__global__ void softmax21_kernel(const float* __restrict__ logits, float* __restrict__ out) {
  const int row = blockIdx.x;
  const int lane = threadIdx.x;
  float v = (lane < 21) ? logits[(size_t)row * 21 + lane] : -INFINITY;
  float m = v;
  for (int o = 16; o > 0; o >>= 1) m = fmaxf(m, __shfl_xor(m, o, 32));
  float e = (lane < 21) ? __expf(v - m) : 0.f;
  float ssum = e;
  for (int o = 16; o > 0; o >>= 1) ssum += __shfl_xor(ssum, o, 32);
  if (lane < 21) out[(size_t)row * 21 + lane] = e / ssum;
}

// =====================================================================================
extern "C" void kernel_launch(void* const* d_in, const int* in_sizes, int n_in,
                              void* d_out, int out_size, void* d_ws, size_t ws_size,
                              hipStream_t stream) {
  (void)in_sizes; (void)n_in; (void)out_size; (void)ws_size;
  const float* x       = (const float*)d_in[0];
  const float* anchors = (const float*)d_in[1];
  // vgg layer t: W=d_in[2+2t], b=d_in[3+2t], t=0..12
  // rpn: W3=d_in[28], b3=d_in[29], Wc=d_in[30], bc=d_in[31], Wr=d_in[32], br=d_in[33]
  // frcn: W1=d_in[34], b1=35, W2=36, b2=37, Wcl=38, bcl=39, Wrg=40, brg=41

  const int ciL[13] = {64, 64, 128, 128, 256, 256, 256, 512, 512, 512, 512, 512, 512};
  const int coL[13] = {64, 128, 128, 256, 256, 256, 512, 512, 512, 512, 512, 512, 512};

  unsigned char* ws = (unsigned char*)d_ws;
  size_t off = 0;
  auto alloc = [&](size_t bytes) -> void* {
    void* p = ws + off;
    off = (off + bytes + 255) & ~(size_t)255;
    return p;
  };

  h16* wf[13];
  for (int i = 0; i < 13; ++i)
    wf[i] = (h16*)alloc((size_t)9 * ciL[i] * coL[i] * sizeof(h16));
  h16* w1h  = (h16*)alloc((size_t)25088 * 4096 * sizeof(h16));
  h16* w2h  = (h16*)alloc((size_t)4096 * 4096 * sizeof(h16));
  h16* wclh = (h16*)alloc((size_t)4096 * 21 * sizeof(h16));
  h16* wrgh = (h16*)alloc((size_t)4096 * 84 * sizeof(h16));
  h16* bufA = (h16*)alloc((size_t)500 * 500 * 64 * sizeof(h16));
  h16* bufB = (h16*)alloc((size_t)500 * 500 * 64 * sizeof(h16));
  float* score  = (float*)alloc((size_t)FHD * FHD * NAA * sizeof(float));
  float* regbuf = (float*)alloc((size_t)FHD * FHD * NAA * 4 * sizeof(float));
  int*   rank   = (int*)alloc((size_t)FHD * FHD * NAA * sizeof(int));
  float* cand   = (float*)alloc((size_t)KTOP * 4 * sizeof(float));
  float* prop   = (float*)alloc((size_t)NPROP * 4 * sizeof(float));
  h16*   zbuf   = (h16*)alloc((size_t)NPROP * 25088 * sizeof(h16));
  h16*   h1     = (h16*)alloc((size_t)NPROP * 4096 * sizeof(h16));
  h16*   h2     = (h16*)alloc((size_t)NPROP * 4096 * sizeof(h16));
  float* logits = (float*)alloc((size_t)NPROP * 21 * sizeof(float));

  // transpose-convert all WMMA weight matrices to [N][K] f16
  auto cvtT = [&](const void* src, h16* dst, int K, int N) {
    long n = (long)K * N;
    int blocks = (int)((n + 256L * 16 - 1) / (256L * 16));
    if (blocks < 1) blocks = 1;
    if (blocks > 8192) blocks = 8192;
    cvt_transpose_kernel<<<blocks, 256, 0, stream>>>((const float*)src, dst, K, N);
  };
  for (int i = 0; i < 12; ++i)
    cvtT(d_in[2 + 2 * (i + 1)], wf[i], 9 * ciL[i], coL[i]);
  cvtT(d_in[28], wf[12], 9 * 512, 512);
  cvtT(d_in[34], w1h, 25088, 4096);
  cvtT(d_in[36], w2h, 4096, 4096);
  cvtT(d_in[38], wclh, 4096, 21);
  cvtT(d_in[40], wrgh, 4096, 84);

  auto conv = [&](const h16* in, const h16* w, const float* bias, h16* out,
                  int H, int Cin, int Cout) {
    dim3 g((H * H + 63) / 64, (Cout + 127) / 128);
    conv3x3_wmma_kernel<<<g, 256, 0, stream>>>(in, w, bias, out, H, Cin, Cout);
  };
  auto pool = [&](const h16* in, h16* out, int Hin, int C) {
    long total = (long)(Hin / 2) * (Hin / 2) * C;
    int blocks = (int)((total + 255) / 256);
    maxpool2_f16_kernel<<<blocks, 256, 0, stream>>>(in, out, Hin, C);
  };
  auto bias = [&](int i) { return (const float*)d_in[i]; };

  // ---- VGG backbone ----
  {
    long total = 500L * 500 * 64;
    conv3x3_c3_kernel<<<(int)((total + 255) / 256), 256, 0, stream>>>(
        x, (const float*)d_in[2], bias(3), bufA, 500);
  }
  conv(bufA, wf[0], bias(5), bufB, 500, 64, 64);
  pool(bufB, bufA, 500, 64);                          // 250x250x64
  conv(bufA, wf[1], bias(7), bufB, 250, 64, 128);
  conv(bufB, wf[2], bias(9), bufA, 250, 128, 128);
  pool(bufA, bufB, 250, 128);                         // 125x125x128
  conv(bufB, wf[3], bias(11), bufA, 125, 128, 256);
  conv(bufA, wf[4], bias(13), bufB, 125, 256, 256);
  conv(bufB, wf[5], bias(15), bufA, 125, 256, 256);
  pool(bufA, bufB, 125, 256);                         // 62x62x256
  conv(bufB, wf[6], bias(17), bufA, 62, 256, 512);
  conv(bufA, wf[7], bias(19), bufB, 62, 512, 512);
  conv(bufB, wf[8], bias(21), bufA, 62, 512, 512);
  pool(bufA, bufB, 62, 512);                          // 31x31x512
  conv(bufB, wf[9],  bias(23), bufA, 31, 512, 512);
  conv(bufA, wf[10], bias(25), bufB, 31, 512, 512);
  conv(bufB, wf[11], bias(27), bufA, 31, 512, 512);   // fmap = bufA
  // ---- RPN ----
  conv(bufA, wf[12], bias(29), bufB, 31, 512, 512);   // h = bufB
  {
    long t1 = (long)FHD * FHD * NAA;
    small_gemm_kernel<<<(int)((t1 + 255) / 256), 256, 0, stream>>>(
        bufB, (const float*)d_in[30], bias(31), score, FHD * FHD, 512, NAA, 1);
    long t2 = (long)FHD * FHD * NAA * 4;
    small_gemm_kernel<<<(int)((t2 + 255) / 256), 256, 0, stream>>>(
        bufB, (const float*)d_in[32], bias(33), regbuf, FHD * FHD, 512, NAA * 4, 0);
  }
  // ---- top-k + decode + NMS ----
  {
    int n = FHD * FHD * NAA;
    topk_rank_kernel<<<(n + 255) / 256, 256, 0, stream>>>(score, rank, n);
    decode_scatter_kernel<<<(n + 255) / 256, 256, 0, stream>>>(rank, regbuf, anchors, cand, n);
    nms_kernel<<<1, 1024, 0, stream>>>(cand, prop);
  }
  // ---- RoI crop + pool ----
  crop_pool_kernel<<<dim3(NPROP, 49), 256, 0, stream>>>(bufA, prop, zbuf);
  // ---- Fast-RCNN head ----
  {
    dim3 g1((NPROP + 63) / 64, 4096 / 128);
    wmma_gemm_kernel<<<g1, 256, 0, stream>>>(zbuf, w1h, bias(35), h1,
                                             NPROP, 4096, 25088, 1, 1);
    wmma_gemm_kernel<<<g1, 256, 0, stream>>>(h1, w2h, bias(37), h2,
                                             NPROP, 4096, 4096, 1, 1);
    dim3 g2((NPROP + 63) / 64, 1);
    wmma_gemm_kernel<<<g2, 256, 0, stream>>>(h2, wclh, bias(39), logits,
                                             NPROP, 21, 4096, 0, 0);
    wmma_gemm_kernel<<<g2, 256, 0, stream>>>(h2, wrgh, bias(41), (float*)d_out,
                                             NPROP, 84, 4096, 0, 0);
    softmax21_kernel<<<NPROP, 32, 0, stream>>>(logits, (float*)d_out + NPROP * 84);
  }
}
